// MultiHeadSelfAttention_56985626083356
// MI455X (gfx1250) — matrix operand order repaired and checked by reading, not compiled, not run on hardware
//
#include <hip/hip_runtime.h>

typedef __attribute__((ext_vector_type(16))) _Float16 v16h;
typedef __attribute__((ext_vector_type(8)))  _Float16 v8h;
typedef __attribute__((ext_vector_type(8)))  float    v8f;
typedef __attribute__((ext_vector_type(4)))  int      v4i;

constexpr int Bb = 2, Ss = 2048, Dm = 1024, Hh = 16, Dh = 64;

// ---- CDNA5 async global->LDS path (ASYNCcnt), guarded for portability ----
#if defined(__has_builtin)
#if __has_builtin(__builtin_amdgcn_global_load_async_to_lds_b128) && \
    __has_builtin(__builtin_amdgcn_s_wait_asynccnt)
#define USE_ASYNC_LDS 1
#endif
#endif
#ifndef USE_ASYNC_LDS
#define USE_ASYNC_LDS 0
#endif

#define AS1 __attribute__((address_space(1)))
#define AS3 __attribute__((address_space(3)))

// copy 16 bytes (8 halves) global -> LDS
static __device__ __forceinline__ void copy16_g2l(const _Float16* g, _Float16* l) {
#if USE_ASYNC_LDS
  __builtin_amdgcn_global_load_async_to_lds_b128(
      (AS1 v4i*)(AS1 void*)(void*)g, (AS3 v4i*)(AS3 void*)(void*)l, 0, 0);
#else
  *(v8h*)l = *(const v8h*)g;
#endif
}
static __device__ __forceinline__ void async_wait_all() {
#if USE_ASYNC_LDS
  __builtin_amdgcn_s_wait_asynccnt(0);
#endif
}

static __device__ __forceinline__ v8f wmma_f16(v16h a, v16h b, v8f c) {
  // D = A(16x32 f16) * B(32x16 f16) + C(16x16 f32)
  return __builtin_amdgcn_wmma_f32_16x16x32_f16(false, a, false, b, (short)0, c,
                                                false, false);
}

static __device__ __forceinline__ void cvt4(v16h& a, int idx, float4 f) {
  a[idx + 0] = (_Float16)f.x;
  a[idx + 1] = (_Float16)f.y;
  a[idx + 2] = (_Float16)f.z;
  a[idx + 3] = (_Float16)f.w;
}

// ---------------------------------------------------------------------------
// Kernel 1: QKV projection with LDS-staged tiles.
//   out = scale * (X @ W^T), written f16 head-split.
//   MODE 0: out[b,h,s,dh]   (Q, K)    MODE 1: out[b,h,dh,s]  (V, transposed)
// Block = 256 thr (8 waves); tile 128(M) x 64(N), K step 32.
// ---------------------------------------------------------------------------
template <int MODE>
__global__ void __launch_bounds__(256)
proj_kernel(const float* __restrict__ X, const float* __restrict__ W,
            _Float16* __restrict__ out, float scale) {
  __shared__ __align__(16) _Float16 xl[128 * 32];  // [m][k] f16, 8 KB
  __shared__ __align__(16) _Float16 wl[64 * 32];   // [n][k] f16, 4 KB

  const int tid  = threadIdx.x;
  const int lane = tid & 31;
  const int wave = tid >> 5;
  const int lrow = lane & 15;
  const int hi   = lane >> 4;
  const int mblk = blockIdx.y * 128;
  const int n0   = blockIdx.x * 64;

  v8f acc[4] = {};

  for (int kk = 0; kk < Dm; kk += 32) {
    // ---- cooperative stage: X tile 128x32 f32 -> f16 LDS ----
    {
      const int xr = tid >> 1;             // 0..127
      const int xc = (tid & 1) * 16;       // 0 or 16
      const float* xs = X + (size_t)(mblk + xr) * Dm + kk + xc;
      __builtin_prefetch(xs + 32, 0, 1);
      v16h xh;
      cvt4(xh, 0,  *(const float4*)(xs + 0));
      cvt4(xh, 4,  *(const float4*)(xs + 4));
      cvt4(xh, 8,  *(const float4*)(xs + 8));
      cvt4(xh, 12, *(const float4*)(xs + 12));
      *(v16h*)&xl[xr * 32 + xc] = xh;
    }
    // ---- cooperative stage: W tile 64x32 f32 -> f16 LDS ----
    {
      const int wr = tid >> 2;             // 0..63
      const int wc = (tid & 3) * 8;        // 0,8,16,24
      const float* ws = W + (size_t)(n0 + wr) * Dm + kk + wc;
      __builtin_prefetch(ws + 32, 0, 1);
      v8h wh;
      float4 g0 = *(const float4*)(ws + 0);
      float4 g1 = *(const float4*)(ws + 4);
      wh[0] = (_Float16)g0.x; wh[1] = (_Float16)g0.y;
      wh[2] = (_Float16)g0.z; wh[3] = (_Float16)g0.w;
      wh[4] = (_Float16)g1.x; wh[5] = (_Float16)g1.y;
      wh[6] = (_Float16)g1.z; wh[7] = (_Float16)g1.w;
      *(v8h*)&wl[wr * 32 + wc] = wh;
    }
    __syncthreads();

    // ---- A fragment from LDS: row = wave*16+lrow, k = e + hi*16 ----
    v16h a;
    {
      const _Float16* p = &xl[(wave * 16 + lrow) * 32 + hi * 16];
      const v8h lo = *(const v8h*)(p);
      const v8h hc = *(const v8h*)(p + 8);
#pragma unroll
      for (int i = 0; i < 8; ++i) { a[i] = lo[i]; a[i + 8] = hc[i]; }
    }
    // ---- B fragments from LDS: col = f*16+lrow, k = e + hi*16 ----
#pragma unroll
    for (int f = 0; f < 4; ++f) {
      const v16h bf = *(const v16h*)&wl[(f * 16 + lrow) * 32 + hi * 16];
      acc[f] = wmma_f16(a, bf, acc[f]);
    }
    __syncthreads();
  }

  // ---- store f16, head-split ----
#pragma unroll
  for (int f = 0; f < 4; ++f) {
    const int n  = n0 + f * 16 + lrow;
    const int h  = n >> 6;
    const int dh = n & 63;
#pragma unroll
    for (int r = 0; r < 8; ++r) {
      const int m = mblk + wave * 16 + r + hi * 8;
      const int b = m >> 11;
      const int s = m & 2047;
      const _Float16 v = (_Float16)(acc[f][r] * scale);
      if (MODE == 0)
        out[((size_t)(b * Hh + h) * Ss + s) * Dh + dh] = v;
      else
        out[((size_t)(b * Hh + h) * Dh + dh) * Ss + s] = v;
    }
  }
}

// ---------------------------------------------------------------------------
// Kernel 2: causal flash attention.
//   q,k: [B,H,S,64] f16;  vT: [B,H,64,S] f16;  out: [B,S,D] f16.
// K/V tiles staged into LDS via the async global->LDS DMA path (ASYNCcnt).
// Per-wave P tile round-trips LDS to convert C-layout -> A-layout.
// ---------------------------------------------------------------------------
__global__ void __launch_bounds__(256)
flash_attn_kernel(const _Float16* __restrict__ Qm, const _Float16* __restrict__ Km,
                  const _Float16* __restrict__ Vt, _Float16* __restrict__ Om) {
  __shared__ __align__(16) _Float16 klds[32 * 64];      // [j][d]   4 KB
  __shared__ __align__(16) _Float16 vlds[64 * 32];      // [d][j]   4 KB
  __shared__ __align__(16) _Float16 plds[8][16 * 32];   // per-wave 8 KB

  const int tid  = threadIdx.x;
  const int lane = tid & 31;
  const int wave = tid >> 5;
  const int lrow = lane & 15;
  const int hi   = lane >> 4;
  const int q0   = blockIdx.x * 128 + wave * 16;
  const int h    = blockIdx.y;
  const int b    = blockIdx.z;
  const size_t base = (size_t)(b * Hh + h) * Ss * Dh;
  const _Float16* qp = Qm + base;
  const _Float16* kp = Km + base;
  const _Float16* vp = Vt + base;   // [64][S]

  // Q as two A-fragments (K-dim = d 0..31 and 32..63)
  v16h qa[2];
#pragma unroll
  for (int s = 0; s < 2; ++s) {
    const _Float16* r = qp + (size_t)(q0 + lrow) * Dh + s * 32 + hi * 16;
    const v8h lo = *(const v8h*)(r);
    const v8h hc = *(const v8h*)(r + 8);
#pragma unroll
    for (int i = 0; i < 8; ++i) { qa[s][i] = lo[i]; qa[s][i + 8] = hc[i]; }
  }

  float rmax[8], rsum[8];
  v8f oacc[4] = {};
#pragma unroll
  for (int r = 0; r < 8; ++r) { rmax[r] = -1e30f; rsum[r] = 0.0f; }

  const int kmax = blockIdx.x * 128 + 112;   // q0 of the last wave in block
  for (int k0 = 0; k0 <= kmax; k0 += 32) {
    // ---- async stage of K tile (32x64) and V^T tile (64x32) into LDS ----
    {
      const int kr = tid >> 3;               // 0..31
      const int kc = (tid & 7) * 8;          // 0..56
      copy16_g2l(kp + (size_t)(k0 + kr) * Dh + kc, &klds[kr * 64 + kc]);
      const int vr = tid >> 2;               // 0..63
      const int vc = (tid & 3) * 8;          // 0..24
      copy16_g2l(vp + (size_t)vr * Ss + k0 + vc, &vlds[vr * 32 + vc]);
    }
    async_wait_all();
    __syncthreads();

    if (k0 <= q0) {   // wave-uniform causal skip; EXEC stays all-1s inside
      // ---- scores: two 16-col subtiles, 2 WMMAs each over Dh=64 ----
      v8f sc[2] = {};
#pragma unroll
      for (int sub = 0; sub < 2; ++sub) {
        const _Float16* kr = &klds[(sub * 16 + lrow) * 64];
        const v16h b0 = *(const v16h*)(kr + hi * 16);        // d 0..31
        const v16h b1 = *(const v16h*)(kr + 32 + hi * 16);   // d 32..63
        sc[sub] = wmma_f16(qa[0], b0, sc[sub]);
        sc[sub] = wmma_f16(qa[1], b1, sc[sub]);
      }
      // ---- causal mask + row max (reduce across 16-lane halves) ----
      float tmax[8];
#pragma unroll
      for (int r = 0; r < 8; ++r) {
        const int row = q0 + r + hi * 8;
        const int c0  = k0 + lrow;
        float s0 = (c0      <= row) ? sc[0][r] : -1e30f;
        float s1 = (c0 + 16 <= row) ? sc[1][r] : -1e30f;
        sc[0][r] = s0; sc[1][r] = s1;
        float mx = fmaxf(s0, s1);
        mx = fmaxf(mx, __shfl_xor(mx, 1, 32));
        mx = fmaxf(mx, __shfl_xor(mx, 2, 32));
        mx = fmaxf(mx, __shfl_xor(mx, 4, 32));
        mx = fmaxf(mx, __shfl_xor(mx, 8, 32));
        tmax[r] = mx;
      }
      // ---- online softmax update, write P tile to per-wave LDS ----
#pragma unroll
      for (int r = 0; r < 8; ++r) {
        const int m = r + hi * 8;
        const float nm    = fmaxf(rmax[r], tmax[r]);
        const float alpha = __expf(rmax[r] - nm);
        const float p0 = __expf(sc[0][r] - nm);
        const float p1 = __expf(sc[1][r] - nm);
        float ps = p0 + p1;
        ps += __shfl_xor(ps, 1, 32);
        ps += __shfl_xor(ps, 2, 32);
        ps += __shfl_xor(ps, 4, 32);
        ps += __shfl_xor(ps, 8, 32);
        rsum[r] = rsum[r] * alpha + ps;
        rmax[r] = nm;
#pragma unroll
        for (int f = 0; f < 4; ++f) oacc[f][r] *= alpha;
        plds[wave][m * 32 + lrow]      = (_Float16)p0;
        plds[wave][m * 32 + 16 + lrow] = (_Float16)p1;
      }
      // ---- reload P in A layout (16x32) ----
      v16h pa;
      {
        const _Float16* pr = &plds[wave][lrow * 32 + hi * 16];
#pragma unroll
        for (int i = 0; i < 8; ++i) { pa[i] = pr[i]; pa[i + 8] = pr[8 + i]; }
      }
      // ---- P @ V: B-frag is a contiguous 32B read of V^T from LDS ----
#pragma unroll
      for (int f = 0; f < 4; ++f) {
        const v16h vb = *(const v16h*)&vlds[(f * 16 + lrow) * 32 + hi * 16];
        oacc[f] = wmma_f16(pa, vb, oacc[f]);
      }
    }
    __syncthreads();   // protect K/V tiles before next iteration's overwrite
  }

  // ---- normalize + store f16 attn output in [B,S,D] (d = h*64 + dh) ----
#pragma unroll
  for (int r = 0; r < 8; ++r) {
    const int m = q0 + r + hi * 8;
    const float inv = 1.0f / rsum[r];
#pragma unroll
    for (int f = 0; f < 4; ++f)
      Om[(size_t)(b * Ss + m) * Dm + h * Dh + f * 16 + lrow] =
          (_Float16)(oacc[f][r] * inv);
  }
}

// ---------------------------------------------------------------------------
// Kernel 3: output projection (f16 A @ f32 W^T -> f32), LDS-staged;
// the f16 A tile uses the async global->LDS path.
// ---------------------------------------------------------------------------
__global__ void __launch_bounds__(256)
out_proj_kernel(const _Float16* __restrict__ A, const float* __restrict__ W,
                float* __restrict__ out) {
  __shared__ __align__(16) _Float16 al[128 * 32];
  __shared__ __align__(16) _Float16 wl[64 * 32];

  const int tid  = threadIdx.x;
  const int lane = tid & 31;
  const int wave = tid >> 5;
  const int lrow = lane & 15;
  const int hi   = lane >> 4;
  const int mblk = blockIdx.y * 128;
  const int n0   = blockIdx.x * 64;

  v8f acc[4] = {};

  for (int kk = 0; kk < Dm; kk += 32) {
    {
      const int xr = tid >> 1;
      const int xc = (tid & 1) * 16;
      copy16_g2l(A + (size_t)(mblk + xr) * Dm + kk + xc, &al[xr * 32 + xc]);
      copy16_g2l(A + (size_t)(mblk + xr) * Dm + kk + xc + 8,
                 &al[xr * 32 + xc + 8]);
    }
    {
      const int wr = tid >> 2;
      const int wc = (tid & 3) * 8;
      const float* ws = W + (size_t)(n0 + wr) * Dm + kk + wc;
      __builtin_prefetch(ws + 32, 0, 1);
      v8h wh;
      float4 g0 = *(const float4*)(ws + 0);
      float4 g1 = *(const float4*)(ws + 4);
      wh[0] = (_Float16)g0.x; wh[1] = (_Float16)g0.y;
      wh[2] = (_Float16)g0.z; wh[3] = (_Float16)g0.w;
      wh[4] = (_Float16)g1.x; wh[5] = (_Float16)g1.y;
      wh[6] = (_Float16)g1.z; wh[7] = (_Float16)g1.w;
      *(v8h*)&wl[wr * 32 + wc] = wh;
    }
    async_wait_all();
    __syncthreads();

    v16h a;
    {
      const _Float16* p = &al[(wave * 16 + lrow) * 32 + hi * 16];
      const v8h lo = *(const v8h*)(p);
      const v8h hc = *(const v8h*)(p + 8);
#pragma unroll
      for (int i = 0; i < 8; ++i) { a[i] = lo[i]; a[i + 8] = hc[i]; }
    }
#pragma unroll
    for (int f = 0; f < 4; ++f) {
      const v16h bf = *(const v16h*)&wl[(f * 16 + lrow) * 32 + hi * 16];
      acc[f] = wmma_f16(a, bf, acc[f]);
    }
    __syncthreads();
  }

#pragma unroll
  for (int f = 0; f < 4; ++f) {
    const int n = n0 + f * 16 + lrow;
#pragma unroll
    for (int r = 0; r < 8; ++r) {
      const int m = mblk + wave * 16 + r + hi * 8;
      out[(size_t)m * Dm + n] = acc[f][r];
    }
  }
}

// ---------------------------------------------------------------------------
extern "C" void kernel_launch(void* const* d_in, const int* in_sizes, int n_in,
                              void* d_out, int out_size, void* d_ws, size_t ws_size,
                              hipStream_t stream) {
  const float* x  = (const float*)d_in[0];
  const float* Qw = (const float*)d_in[1];
  const float* Kw = (const float*)d_in[2];
  const float* Vw = (const float*)d_in[3];
  const float* Ow = (const float*)d_in[4];
  float* out = (float*)d_out;

  const size_t elems = (size_t)Bb * Ss * Dm;   // 4 Mi elements
  _Float16* qb = (_Float16*)d_ws;
  _Float16* kb = qb + elems;
  _Float16* vb = kb + elems;   // holds V^T per head: [B,H,Dh,S]
  _Float16* ab = vb + elems;

  const dim3 gemm_grid(Dm / 64, (Bb * Ss) / 128);     // 16 x 32
  const dim3 attn_grid(Ss / 128, Hh, Bb);             // 16 x 16 x 2
  const dim3 blk(256);

  const float scale = 0.125f;   // 1/sqrt(Dh=64), folded into q
  proj_kernel<0><<<gemm_grid, blk, 0, stream>>>(x, Qw, qb, scale);
  proj_kernel<0><<<gemm_grid, blk, 0, stream>>>(x, Kw, kb, 1.0f);
  proj_kernel<1><<<gemm_grid, blk, 0, stream>>>(x, Vw, vb, 1.0f);
  flash_attn_kernel<<<attn_grid, blk, 0, stream>>>(qb, kb, vb, ab);
  out_proj_kernel<<<gemm_grid, blk, 0, stream>>>(ab, Ow, out);
}
